// PGNNLayers_69398081569452
// MI455X (gfx1250) — compile-verified, matrix operand order checked by reading, not run on hardware
//
#include <hip/hip_runtime.h>
#include <hip/hip_bf16.h>

#define NN 16384
#define LL 50

#if defined(__has_builtin)
#  if __has_builtin(__builtin_amdgcn_global_load_async_to_lds_b128) && \
      __has_builtin(__builtin_amdgcn_s_wait_asynccnt)
#    define USE_ASYNC_GATHER 1
#  endif
#endif

typedef __attribute__((ext_vector_type(16))) __bf16 v16bf;
typedef __attribute__((ext_vector_type(8)))  float  v8f;
typedef __attribute__((ext_vector_type(4)))  int    v4i;
typedef __attribute__((address_space(1)))    v4i    gv4i;
typedef __attribute__((address_space(3)))    v4i    lv4i;

__device__ __forceinline__ __bf16 f2bf(float f) { return (__bf16)f; }

__device__ __forceinline__ int frag_k(int v, int h) {
  return ((v < 4) ? 2 * v : 2 * v + 8) + 8 * h;
}

// 16x32 bf16 fragment from f32 row-major [16 x ld] tile (vector dim = rows for A,
// = output columns for B since B[k][n] = W[n][k] with W row-major).
__device__ __forceinline__ v16bf load_frag_f32(const float* p, int ld, int lane) {
  v16bf f; int m = lane & 15, h = lane >> 4;
#pragma unroll
  for (int v = 0; v < 8; ++v) {
    int k = frag_k(v, h);
    f[2 * v]     = f2bf(p[m * ld + k]);
    f[2 * v + 1] = f2bf(p[m * ld + k + 1]);
  }
  return f;
}

__device__ __forceinline__ v16bf load_frag_bf16(const __bf16* p, int ld, int lane) {
  v16bf f; int m = lane & 15, h = lane >> 4;
#pragma unroll
  for (int v = 0; v < 8; ++v) {
    int k = frag_k(v, h);
    f[2 * v]     = p[m * ld + k];
    f[2 * v + 1] = p[m * ld + k + 1];
  }
  return f;
}

// per-lane row pointer variant (clamped attention rows)
__device__ __forceinline__ v16bf load_frag_hrow(const __bf16* rowp, int kbase, int lane) {
  v16bf f; int h = lane >> 4;
#pragma unroll
  for (int v = 0; v < 8; ++v) {
    int k = kbase + frag_k(v, h);
    f[2 * v]     = rowp[k];
    f[2 * v + 1] = rowp[k + 1];
  }
  return f;
}

// concat of two row-major [M,64] f32 matrices as a [M,128] A operand
__device__ __forceinline__ v16bf load_frag_cat(const float* A0, const float* A1,
                                               int row0, int kbase, int lane) {
  v16bf f; int m = row0 + (lane & 15), h = lane >> 4;
#pragma unroll
  for (int v = 0; v < 8; ++v) {
    int k = kbase + frag_k(v, h);
    const float* s = (k < 64) ? (A0 + m * 64 + k) : (A1 + m * 64 + (k - 64));
    f[2 * v]     = f2bf(s[0]);
    f[2 * v + 1] = f2bf(s[1]);
  }
  return f;
}

__device__ __forceinline__ v8f wmma_bf(v16bf a, v16bf b, v8f c) {
  return __builtin_amdgcn_wmma_f32_16x16x32_bf16(false, a, false, b, (short)0, c,
                                                 false, false);
}

// ---------------- C[M,64] = A[M,64] @ W[64,64]^T ----------------
__global__ __launch_bounds__(256)
void k_gemm64(const float* __restrict__ A, const float* __restrict__ W,
              float* __restrict__ C, int M) {
  __shared__ __bf16 sW[64 * 64];
  for (int i = threadIdx.x; i < 64 * 64; i += 256) sW[i] = f2bf(W[i]);
  __syncthreads();
  int wave = threadIdx.x >> 5, lane = threadIdx.x & 31;
  int row0 = (blockIdx.x * 8 + wave) * 16;
  if (row0 >= M) return;
  v16bf a0 = load_frag_f32(A + row0 * 64, 64, lane);
  v16bf a1 = load_frag_f32(A + row0 * 64 + 32, 64, lane);
#pragma unroll
  for (int nt = 0; nt < 4; ++nt) {
    v8f acc{};
    acc = wmma_bf(a0, load_frag_bf16(sW + nt * 16 * 64, 64, lane), acc);
    acc = wmma_bf(a1, load_frag_bf16(sW + nt * 16 * 64 + 32, 64, lane), acc);
    int n = nt * 16 + (lane & 15), mh = (lane >> 4) * 8;
#pragma unroll
    for (int r = 0; r < 8; ++r) C[(row0 + mh + r) * 64 + n] = acc[r];
  }
}

// ------------- C[M,64] = [A0|A1][M,128] @ W[64,128]^T, optional tanh -------------
__global__ __launch_bounds__(256)
void k_gemm_cat128(const float* __restrict__ A0, const float* __restrict__ A1,
                   const float* __restrict__ W, float* __restrict__ C,
                   int M, int do_tanh) {
  __shared__ __bf16 sW[64 * 128];
  for (int i = threadIdx.x; i < 64 * 128; i += 256) sW[i] = f2bf(W[i]);
  __syncthreads();
  int wave = threadIdx.x >> 5, lane = threadIdx.x & 31;
  int row0 = (blockIdx.x * 8 + wave) * 16;
  if (row0 >= M) return;
  v16bf a[4];
#pragma unroll
  for (int ks = 0; ks < 4; ++ks) a[ks] = load_frag_cat(A0, A1, row0, ks * 32, lane);
#pragma unroll
  for (int nt = 0; nt < 4; ++nt) {
    v8f acc{};
#pragma unroll
    for (int ks = 0; ks < 4; ++ks)
      acc = wmma_bf(a[ks], load_frag_bf16(sW + nt * 16 * 128 + ks * 32, 128, lane), acc);
    int n = nt * 16 + (lane & 15), mh = (lane >> 4) * 8;
#pragma unroll
    for (int r = 0; r < 8; ++r) {
      float vv = acc[r];
      if (do_tanh) vv = tanhf(vv);
      C[(row0 + mh + r) * 64 + n] = vv;
    }
  }
}

// ------------- per-node stable argsort of times; rrank = L-1-rank; argmax -------------
__global__ __launch_bounds__(256)
void k_sort(const int* __restrict__ times, int* __restrict__ perm,
            int* __restrict__ rrank, int* __restrict__ lastIdx) {
  int n = blockIdx.x * blockDim.x + threadIdx.x;
  if (n >= NN) return;
  int t[LL], p[LL];
  const int* tp = times + n * LL;
  for (int i = 0; i < LL; ++i) { t[i] = tp[i]; p[i] = i; }
  for (int i = 1; i < LL; ++i) {
    int tv = t[i], pv = p[i], j = i - 1;
    while (j >= 0 && t[j] > tv) { t[j + 1] = t[j]; p[j + 1] = p[j]; --j; }
    t[j + 1] = tv; p[j + 1] = pv;
  }
  for (int k = 0; k < LL; ++k) {
    perm[n * LL + k] = p[k];
    rrank[n * LL + p[k]] = LL - 1 - k;
  }
  int bi = 0, bt = tp[0];
  for (int i = 1; i < LL; ++i) if (tp[i] > bt) { bt = tp[i]; bi = i; }
  lastIdx[n] = bi;
}

// ------------- fused gather + GRU (one direction); 4 waves x 16 nodes/WG -------------
__global__ __launch_bounds__(128)
void k_gru(const float* __restrict__ srcH, const int* __restrict__ nbrIdx,
           const int* __restrict__ perm,
           const float* __restrict__ wih, const float* __restrict__ whh,
           const float* __restrict__ bih, const float* __restrict__ bhh,
           __bf16* __restrict__ hcat, int dir) {
  __shared__ __bf16 sWih[192 * 64];
  __shared__ __bf16 sWhh[192 * 64];
  __shared__ float  sBih[192], sBhh[192];
  __shared__ float  sX[4][16 * 64];
  __shared__ __bf16 sH[4][16 * 64];
  __shared__ int    sSrc[4][16];
  int tid = threadIdx.x;
  for (int i = tid; i < 192 * 64; i += 128) { sWih[i] = f2bf(wih[i]); sWhh[i] = f2bf(whh[i]); }
  for (int i = tid; i < 192; i += 128) { sBih[i] = bih[i]; sBhh[i] = bhh[i]; }
  int wave = tid >> 5, lane = tid & 31;
  __bf16* mH = sH[wave];
  float*  mX = sX[wave];
  for (int i = lane; i < 16 * 64; i += 32) mH[i] = (__bf16)0.f;
  __syncthreads();
  int nodeBase = blockIdx.x * 64 + wave * 16;
  const float4* src4 = (const float4*)srcH;
  for (int step = 0; step < LL; ++step) {
    int pos = dir ? (LL - 1 - step) : step;
    // per-row gather index (once per row) + prefetch of next step's rows
    if (lane < 16) {
      int node = nodeBase + lane;
      sSrc[wave][lane] = nbrIdx[node * LL + perm[node * LL + pos]];
      if (step + 1 < LL) {
        int pos2 = dir ? (LL - 2 - step) : (step + 1);
        int srow2 = nbrIdx[node * LL + perm[node * LL + pos2]];
        __builtin_prefetch(srcH + (size_t)srow2 * 64, 0, 3);
      }
    }
    __builtin_amdgcn_wave_barrier();
    // gather x tile (16 rows x 16 float4) into LDS
#ifdef USE_ASYNC_GATHER
    for (int e = lane; e < 16 * 16; e += 32) {
      int r = e >> 4, c4 = e & 15;
      __builtin_amdgcn_global_load_async_to_lds_b128(
          (gv4i*)(srcH + (size_t)sSrc[wave][r] * 64 + c4 * 4),
          (lv4i*)(mX + r * 64 + c4 * 4),
          0, 0);
    }
    __builtin_amdgcn_s_wait_asynccnt(0);
#else
    for (int e = lane; e < 16 * 16; e += 32) {
      int r = e >> 4, c4 = e & 15;
      float4 vv = src4[(size_t)sSrc[wave][r] * 16 + c4];
      ((float4*)mX)[r * 16 + c4] = vv;
    }
#endif
    __syncthreads();
    // A fragments for this step (x and previous h), shared across gate groups
    v16bf ax0 = load_frag_f32(mX, 64, lane);
    v16bf ax1 = load_frag_f32(mX + 32, 64, lane);
    v16bf ah0 = load_frag_bf16(mH, 64, lane);
    v16bf ah1 = load_frag_bf16(mH + 32, 64, lane);
    float hnew[4][8];
#pragma unroll
    for (int g = 0; g < 4; ++g) {
      v8f ir{}, iz{}, in_{}, hr{}, hz{}, hn{};
      ir  = wmma_bf(ax0, load_frag_bf16(sWih + (g)     * 16 * 64,      64, lane), ir);
      ir  = wmma_bf(ax1, load_frag_bf16(sWih + (g)     * 16 * 64 + 32, 64, lane), ir);
      iz  = wmma_bf(ax0, load_frag_bf16(sWih + (g + 4) * 16 * 64,      64, lane), iz);
      iz  = wmma_bf(ax1, load_frag_bf16(sWih + (g + 4) * 16 * 64 + 32, 64, lane), iz);
      in_ = wmma_bf(ax0, load_frag_bf16(sWih + (g + 8) * 16 * 64,      64, lane), in_);
      in_ = wmma_bf(ax1, load_frag_bf16(sWih + (g + 8) * 16 * 64 + 32, 64, lane), in_);
      hr  = wmma_bf(ah0, load_frag_bf16(sWhh + (g)     * 16 * 64,      64, lane), hr);
      hr  = wmma_bf(ah1, load_frag_bf16(sWhh + (g)     * 16 * 64 + 32, 64, lane), hr);
      hz  = wmma_bf(ah0, load_frag_bf16(sWhh + (g + 4) * 16 * 64,      64, lane), hz);
      hz  = wmma_bf(ah1, load_frag_bf16(sWhh + (g + 4) * 16 * 64 + 32, 64, lane), hz);
      hn  = wmma_bf(ah0, load_frag_bf16(sWhh + (g + 8) * 16 * 64,      64, lane), hn);
      hn  = wmma_bf(ah1, load_frag_bf16(sWhh + (g + 8) * 16 * 64 + 32, 64, lane), hn);
      int cg = g * 16 + (lane & 15);
      int mh = (lane >> 4) * 8;
#pragma unroll
      for (int r = 0; r < 8; ++r) {
        float irv = ir[r]  + sBih[cg],        hrv = hr[r] + sBhh[cg];
        float izv = iz[r]  + sBih[64 + cg],   hzv = hz[r] + sBhh[64 + cg];
        float inv = in_[r] + sBih[128 + cg],  hnv = hn[r] + sBhh[128 + cg];
        float rg = 1.f / (1.f + expf(-(irv + hrv)));
        float zg = 1.f / (1.f + expf(-(izv + hzv)));
        float ng = tanhf(inv + rg * hnv);
        float hold = (float)mH[(mh + r) * 64 + cg];
        hnew[g][r] = (1.f - zg) * ng + zg * hold;
      }
    }
    __syncthreads();
    int mh = (lane >> 4) * 8;
#pragma unroll
    for (int g = 0; g < 4; ++g) {
      int cg = g * 16 + (lane & 15);
#pragma unroll
      for (int r = 0; r < 8; ++r) {
        int m = mh + r;
        __bf16 hb = f2bf(hnew[g][r]);
        mH[m * 64 + cg] = hb;
        hcat[((size_t)(nodeBase + m) * LL + pos) * 128 + dir * 64 + cg] = hb;
      }
    }
    __syncthreads();
  }
}

// ------------- attention logits: e[n,p] = <te[rrank]+seq, dstH[n]>/8 -------------
__global__ __launch_bounds__(128)
void k_attn_e(const __bf16* __restrict__ hcat, const float* __restrict__ lin,
              const float* __restrict__ te, const int* __restrict__ rrank,
              const float* __restrict__ dstH, float* __restrict__ e) {
  __shared__ __bf16 sLin[64 * 128];
  for (int i = threadIdx.x; i < 64 * 128; i += 128) sLin[i] = f2bf(lin[i]);
  __syncthreads();
  int wave = threadIdx.x >> 5, lane = threadIdx.x & 31;
  int n = blockIdx.x * 4 + wave;
  int half = lane >> 4;
#pragma unroll
  for (int c0 = 0; c0 < 64; c0 += 16) {
    int pm = c0 + (lane & 15); if (pm > LL - 1) pm = LL - 1;
    const __bf16* rowp = hcat + ((size_t)n * LL + pm) * 128;
    v16bf a[4];
#pragma unroll
    for (int ks = 0; ks < 4; ++ks) a[ks] = load_frag_hrow(rowp, ks * 32, lane);
    float part[8];
#pragma unroll
    for (int r = 0; r < 8; ++r) part[r] = 0.f;
#pragma unroll
    for (int nt = 0; nt < 4; ++nt) {
      v8f acc{};
#pragma unroll
      for (int ks = 0; ks < 4; ++ks)
        acc = wmma_bf(a[ks], load_frag_bf16(sLin + nt * 16 * 128 + ks * 32, 128, lane), acc);
      int col = nt * 16 + (lane & 15);
      float dh = dstH[n * 64 + col];
#pragma unroll
      for (int r = 0; r < 8; ++r) {
        int prow = c0 + half * 8 + r;
        if (prow < LL) {
          int rr = rrank[n * LL + prow];
          part[r] += (acc[r] + te[rr * 64 + col]) * dh;
        }
      }
    }
#pragma unroll
    for (int off = 1; off < 16; off <<= 1)
#pragma unroll
      for (int r = 0; r < 8; ++r) part[r] += __shfl_xor(part[r], off, 32);
    if ((lane & 15) == 0) {
#pragma unroll
      for (int r = 0; r < 8; ++r) {
        int prow = c0 + half * 8 + r;
        if (prow < LL) e[n * LL + prow] = part[r] * 0.125f;
      }
    }
  }
}

__global__ __launch_bounds__(256)
void k_softmax(float* __restrict__ e) {
  int n = blockIdx.x * blockDim.x + threadIdx.x;
  if (n >= NN) return;
  float* ep = e + n * LL;
  float mx = ep[0];
  for (int i = 1; i < LL; ++i) mx = fmaxf(mx, ep[i]);
  float v[LL], s = 0.f;
  for (int i = 0; i < LL; ++i) { v[i] = expf(ep[i] - mx); s += v[i]; }
  float inv = 1.f / s;
  for (int i = 0; i < LL; ++i) ep[i] = v[i] * inv;
}

// ------------- hid[n] = sum_p alpha * (seq + te_k[rrank]);  lastem gather -------------
__global__ __launch_bounds__(128)
void k_attn_hid(const __bf16* __restrict__ hcat, const float* __restrict__ lin,
                const float* __restrict__ tek, const int* __restrict__ rrank,
                const float* __restrict__ alpha, const int* __restrict__ lastIdx,
                const int* __restrict__ nbrIdx, const float* __restrict__ srcH,
                float* __restrict__ hid, float* __restrict__ lastem) {
  __shared__ __bf16 sLin[64 * 128];
  for (int i = threadIdx.x; i < 64 * 128; i += 128) sLin[i] = f2bf(lin[i]);
  __syncthreads();
  int wave = threadIdx.x >> 5, lane = threadIdx.x & 31;
  int n = blockIdx.x * 4 + wave;
  int half = lane >> 4;
  float acc4[4] = {0.f, 0.f, 0.f, 0.f};
#pragma unroll
  for (int c0 = 0; c0 < 64; c0 += 16) {
    int pm = c0 + (lane & 15); if (pm > LL - 1) pm = LL - 1;
    const __bf16* rowp = hcat + ((size_t)n * LL + pm) * 128;
    v16bf a[4];
#pragma unroll
    for (int ks = 0; ks < 4; ++ks) a[ks] = load_frag_hrow(rowp, ks * 32, lane);
#pragma unroll
    for (int nt = 0; nt < 4; ++nt) {
      v8f acc{};
#pragma unroll
      for (int ks = 0; ks < 4; ++ks)
        acc = wmma_bf(a[ks], load_frag_bf16(sLin + nt * 16 * 128 + ks * 32, 128, lane), acc);
      int col = nt * 16 + (lane & 15);
#pragma unroll
      for (int r = 0; r < 8; ++r) {
        int prow = c0 + half * 8 + r;
        if (prow < LL) {
          int rr = rrank[n * LL + prow];
          acc4[nt] += alpha[n * LL + prow] * (acc[r] + tek[rr * 64 + col]);
        }
      }
    }
  }
#pragma unroll
  for (int nt = 0; nt < 4; ++nt) acc4[nt] += __shfl_xor(acc4[nt], 16, 32);
  if (lane < 16) {
#pragma unroll
    for (int nt = 0; nt < 4; ++nt) hid[n * 64 + nt * 16 + lane] = acc4[nt];
  }
  int srow = nbrIdx[n * LL + lastIdx[n]];
  for (int c = lane; c < 64; c += 32) lastem[n * 64 + c] = srcH[srow * 64 + c];
}

extern "C" void kernel_launch(void* const* d_in, const int* in_sizes, int n_in,
                              void* d_out, int out_size, void* d_ws, size_t ws_size,
                              hipStream_t stream) {
  (void)in_sizes; (void)n_in; (void)out_size; (void)ws_size;
  const float* user_feat = (const float*)d_in[0];
  const float* item_feat = (const float*)d_in[1];
  const int*   u_idx     = (const int*)d_in[2];
  const int*   u_time    = (const int*)d_in[3];
  const int*   i_idx     = (const int*)d_in[4];
  const int*   i_time    = (const int*)d_in[5];
  const float* Wu        = (const float*)d_in[6];
  const float* Wi        = (const float*)d_in[7];
  const float* gru_u_wih = (const float*)d_in[8];
  const float* gru_u_whh = (const float*)d_in[9];
  const float* gru_u_b   = (const float*)d_in[10];
  const float* gru_i_wih = (const float*)d_in[11];
  const float* gru_i_whh = (const float*)d_in[12];
  const float* gru_i_b   = (const float*)d_in[13];
  const float* lin_u     = (const float*)d_in[14];
  const float* lin_i     = (const float*)d_in[15];
  const float* u_te      = (const float*)d_in[16];
  const float* u_tek     = (const float*)d_in[17];
  const float* i_te      = (const float*)d_in[18];
  const float* i_tek     = (const float*)d_in[19];
  const float* gate_u    = (const float*)d_in[20];
  const float* gate_i    = (const float*)d_in[21];
  const float* upd_u     = (const float*)d_in[22];
  const float* upd_i     = (const float*)d_in[23];
  float* out = (float*)d_out;

  char* ws = (char*)d_ws;
  const size_t NB = (size_t)NN * 64 * sizeof(float);  // 4 MiB
  float*  uh     = (float*)(ws + 0 * NB);
  float*  ih     = (float*)(ws + 1 * NB);
  float*  newU   = (float*)(ws + 2 * NB);
  float*  newI   = (float*)(ws + 3 * NB);
  float*  hid    = (float*)(ws + 4 * NB);
  float*  lastem = (float*)(ws + 5 * NB);
  int*    perm   = (int*)  (ws + 6 * NB);
  int*    rrank  = (int*)  (ws + 7 * NB);
  float*  evec   = (float*)(ws + 8 * NB);
  int*    lastI  = (int*)  (ws + 9 * NB);
  __bf16* hcat   = (__bf16*)(ws + 10 * NB);  // [N,50,128] bf16 ~= 210 MB

  const int gGemm = NN / 128;  // 8 waves x 16 rows per block

  k_gemm64<<<gGemm, 256, 0, stream>>>(user_feat, Wu, uh, NN);
  k_gemm64<<<gGemm, 256, 0, stream>>>(item_feat, Wi, ih, NN);

  // ---- user side: dst uh, src ih ----
  k_sort<<<NN / 256, 256, 0, stream>>>(u_time, perm, rrank, lastI);
  k_gru<<<NN / 64, 128, 0, stream>>>(ih, u_idx, perm, gru_u_wih, gru_u_whh,
                                     gru_u_b, gru_u_b + 192, hcat, 0);
  k_gru<<<NN / 64, 128, 0, stream>>>(ih, u_idx, perm, gru_u_wih + 192 * 64,
                                     gru_u_whh + 192 * 64, gru_u_b + 384,
                                     gru_u_b + 384 + 192, hcat, 1);
  k_attn_e<<<NN / 4, 128, 0, stream>>>(hcat, lin_u, u_te, rrank, uh, evec);
  k_softmax<<<NN / 256, 256, 0, stream>>>(evec);
  k_attn_hid<<<NN / 4, 128, 0, stream>>>(hcat, lin_u, u_tek, rrank, evec, lastI,
                                         u_idx, ih, hid, lastem);
  k_gemm_cat128<<<gGemm, 256, 0, stream>>>(hid, lastem, gate_u, newU, NN, 0);

  // ---- item side: dst ih, src uh ----
  k_sort<<<NN / 256, 256, 0, stream>>>(i_time, perm, rrank, lastI);
  k_gru<<<NN / 64, 128, 0, stream>>>(uh, i_idx, perm, gru_i_wih, gru_i_whh,
                                     gru_i_b, gru_i_b + 192, hcat, 0);
  k_gru<<<NN / 64, 128, 0, stream>>>(uh, i_idx, perm, gru_i_wih + 192 * 64,
                                     gru_i_whh + 192 * 64, gru_i_b + 384,
                                     gru_i_b + 384 + 192, hcat, 1);
  k_attn_e<<<NN / 4, 128, 0, stream>>>(hcat, lin_i, i_te, rrank, ih, evec);
  k_softmax<<<NN / 256, 256, 0, stream>>>(evec);
  k_attn_hid<<<NN / 4, 128, 0, stream>>>(hcat, lin_i, i_tek, rrank, evec, lastI,
                                         i_idx, uh, hid, lastem);
  k_gemm_cat128<<<gGemm, 256, 0, stream>>>(hid, lastem, gate_i, newI, NN, 0);

  // ---- final updates: tanh([new, feat] @ upd.T) ----
  k_gemm_cat128<<<gGemm, 256, 0, stream>>>(newU, user_feat, upd_u, out, NN, 1);
  k_gemm_cat128<<<gGemm, 256, 0, stream>>>(newI, item_feat, upd_i,
                                           out + (size_t)NN * 64, NN, 1);
}
